// PointTransformerBackbone_56916906606938
// MI455X (gfx1250) — compile-verified, matrix operand order checked by reading, not run on hardware
//
#include <hip/hip_runtime.h>
#include <hip/hip_bf16.h>

// ---------------------------------------------------------------------------
// PointTransformer backbone for gfx1250 (MI455X).
// All GEMMs go through v_wmma_f32_16x16x32_f16 (f16 inputs, f32 accumulate).
// GEMM tiles are always full (M,N multiples of 16 in this network); the main
// K loop uses unconditional 128-bit loads so no EXEC-mask branching occurs.
// ---------------------------------------------------------------------------

typedef __attribute__((ext_vector_type(16))) _Float16 v16h;
typedef __attribute__((ext_vector_type(8)))  float    v8f;

static __device__ __forceinline__ float relu_f(float x) { return fmaxf(x, 0.f); }

// ------------------------------ GEMM (WMMA) --------------------------------
// Y[M,N] = act( X[M,K] @ W[N,K]^T + bias[N] )    (act: 0 = none, 1 = relu)
// One wave per 16x16 output tile; 8 waves per block.
// REQUIRES: M % 16 == 0, N % 16 == 0 (always true in this network).
__global__ __launch_bounds__(256) void k_gemm_wmma(
    const float* __restrict__ X, const float* __restrict__ W,
    const float* __restrict__ bias, float* __restrict__ Y,
    int M, int N, int K, int act)
{
    int wave = threadIdx.x >> 5;
    int lane = threadIdx.x & 31;
    int tilesN = N >> 4;
    int tilesM = M >> 4;
    int tile = blockIdx.x * 8 + wave;
    if (tile >= tilesM * tilesN) return;          // wave-uniform exit
    int tm = (tile / tilesN) << 4;
    int tn = (tile % tilesN) << 4;

    int r  = lane & 15;                           // row-in-tile (A) / col (B)
    int hi = lane >> 4;                           // half-wave select
    const float* Xr = X + (size_t)(tm + r) * K;   // A row for this lane
    const float* Wr = W + (size_t)(tn + r) * K;   // B row (output col)

    v8f acc = {};
    // 16-bit A 16x32 layout: lanes 0-15 hold K=0..7 (regs 0..3) and K=16..23
    // (regs 4..7); lanes 16-31 hold K=8..15 and 24..31.  B mirrors (N=lane%16).
    int Kmain = K & ~31;
    for (int k0 = 0; k0 < Kmain; k0 += 32) {
        int base = k0 + (hi << 3);
        float4 a0 = *(const float4*)(Xr + base);
        float4 a1 = *(const float4*)(Xr + base + 4);
        float4 a2 = *(const float4*)(Xr + base + 16);
        float4 a3 = *(const float4*)(Xr + base + 20);
        float4 b0 = *(const float4*)(Wr + base);
        float4 b1 = *(const float4*)(Wr + base + 4);
        float4 b2 = *(const float4*)(Wr + base + 16);
        float4 b3 = *(const float4*)(Wr + base + 20);
        v16h a, b;
        a[0]  = (_Float16)a0.x; a[1]  = (_Float16)a0.y; a[2]  = (_Float16)a0.z; a[3]  = (_Float16)a0.w;
        a[4]  = (_Float16)a1.x; a[5]  = (_Float16)a1.y; a[6]  = (_Float16)a1.z; a[7]  = (_Float16)a1.w;
        a[8]  = (_Float16)a2.x; a[9]  = (_Float16)a2.y; a[10] = (_Float16)a2.z; a[11] = (_Float16)a2.w;
        a[12] = (_Float16)a3.x; a[13] = (_Float16)a3.y; a[14] = (_Float16)a3.z; a[15] = (_Float16)a3.w;
        b[0]  = (_Float16)b0.x; b[1]  = (_Float16)b0.y; b[2]  = (_Float16)b0.z; b[3]  = (_Float16)b0.w;
        b[4]  = (_Float16)b1.x; b[5]  = (_Float16)b1.y; b[6]  = (_Float16)b1.z; b[7]  = (_Float16)b1.w;
        b[8]  = (_Float16)b2.x; b[9]  = (_Float16)b2.y; b[10] = (_Float16)b2.z; b[11] = (_Float16)b2.w;
        b[12] = (_Float16)b3.x; b[13] = (_Float16)b3.y; b[14] = (_Float16)b3.z; b[15] = (_Float16)b3.w;
        acc = __builtin_amdgcn_wmma_f32_16x16x32_f16(
            false, a, false, b, (short)0, acc, false, false);
    }
    if (Kmain < K) {                              // K tail (only K=3 case)
        v16h a, b;
#pragma unroll
        for (int e = 0; e < 16; ++e) {
            int k  = Kmain + (hi << 3) + ((e < 8) ? e : (e + 8));
            int kc = (k < K) ? k : 0;             // clamped, unconditional load
            float av = Xr[kc];
            float bv = Wr[kc];
            a[e] = (_Float16)((k < K) ? av : 0.f);
            b[e] = (_Float16)((k < K) ? bv : 0.f);
        }
        acc = __builtin_amdgcn_wmma_f32_16x16x32_f16(
            false, a, false, b, (short)0, acc, false, false);
    }
    // C/D layout: reg v -> M = v + 8*hi ; N = lane%16
    float bval = bias ? bias[tn + r] : 0.f;
#pragma unroll
    for (int v = 0; v < 8; ++v) {
        float val = acc[v] + bval;
        if (act) val = relu_f(val);
        Y[(size_t)(tm + v + (hi << 3)) * N + (tn + r)] = val;
    }
}

// ------------------------------ LayerNorm ----------------------------------
// One wave per row. out = (x-mu)*rsqrt(var+1e-5)*g + b ; optional relu; optional +add
__global__ __launch_bounds__(32) void k_ln(
    const float* __restrict__ in, float* __restrict__ out,
    const float* __restrict__ g, const float* __restrict__ b,
    int C, int relu, const float* __restrict__ add)
{
    int row  = blockIdx.x;
    int lane = threadIdx.x;
    const float* r = in + (size_t)row * C;
    float s = 0.f, s2 = 0.f;
    for (int c = lane; c < C; c += 32) { float v = r[c]; s += v; s2 += v * v; }
#pragma unroll
    for (int m = 16; m; m >>= 1) { s += __shfl_xor(s, m, 32); s2 += __shfl_xor(s2, m, 32); }
    float mu  = s / (float)C;
    float var = s2 / (float)C - mu * mu;
    float inv = rsqrtf(fmaxf(var, 0.f) + 1e-5f);
    float* o = out + (size_t)row * C;
    for (int c = lane; c < C; c += 32) {
        float v = (r[c] - mu) * inv * g[c] + b[c];
        if (relu) v = relu_f(v);
        if (add)  v += add[(size_t)row * C + c];
        o[c] = v;
    }
}

// ------------------------------ KNN ----------------------------------------
// One thread per query; insertion list of KK smallest squared distances.
template <int KK, bool EXCL>
__global__ __launch_bounds__(256) void k_knn(
    const float* __restrict__ q, const float* __restrict__ bp,
    int nq, int nb, int* __restrict__ out)
{
    int i = blockIdx.x * blockDim.x + threadIdx.x;
    if (i >= nq) return;
    float qx = q[3 * i], qy = q[3 * i + 1], qz = q[3 * i + 2];
    float bd[KK]; int bi[KK];
#pragma unroll
    for (int k = 0; k < KK; ++k) { bd[k] = 3.0e38f; bi[k] = 0; }
    for (int j = 0; j < nb; ++j) {
        if (EXCL && j == i) continue;
        float dx = qx - bp[3 * j], dy = qy - bp[3 * j + 1], dz = qz - bp[3 * j + 2];
        float d = dx * dx + dy * dy + dz * dz;
        if (d < bd[KK - 1]) {
            int k = KK - 1;
            while (k > 0 && bd[k - 1] > d) { bd[k] = bd[k - 1]; bi[k] = bi[k - 1]; --k; }
            bd[k] = d; bi[k] = j;
        }
    }
#pragma unroll
    for (int k = 0; k < KK; ++k) out[(size_t)i * KK + k] = bi[k];
}

// ------------------------------ FPS ----------------------------------------
// Single block; sequential farthest-point sampling with LDS argmax reduction.
__global__ __launch_bounds__(256) void k_fps(
    const float* __restrict__ pos, int n, int m, int* __restrict__ out)
{
    __shared__ float mind[4096];
    __shared__ float redv[256];
    __shared__ int   redi[256];
    int t = threadIdx.x;
    for (int j = t; j < n; j += 256) mind[j] = 3.0e38f;
    __syncthreads();
    int cur = 0;
    for (int s = 0; s < m; ++s) {
        if (t == 0) out[s] = cur;
        float cx = pos[3 * cur], cy = pos[3 * cur + 1], cz = pos[3 * cur + 2];
        float best = -1.f; int bidx = 0x7fffffff;
        for (int j = t; j < n; j += 256) {
            float dx = pos[3 * j] - cx, dy = pos[3 * j + 1] - cy, dz = pos[3 * j + 2] - cz;
            float d = dx * dx + dy * dy + dz * dz;
            float nd = fminf(mind[j], d);
            mind[j] = nd;
            if (nd > best) { best = nd; bidx = j; }
        }
        redv[t] = best; redi[t] = bidx;
        __syncthreads();
        for (int off = 128; off; off >>= 1) {
            if (t < off) {
                float ov = redv[t + off]; int oi = redi[t + off];
                if (ov > redv[t] || (ov == redv[t] && oi < redi[t])) { redv[t] = ov; redi[t] = oi; }
            }
            __syncthreads();
        }
        cur = redi[0];
        __syncthreads();
    }
}

// ------------------------------ small helpers ------------------------------
__global__ __launch_bounds__(256) void k_gather(
    const float* __restrict__ in, const int* __restrict__ idx,
    float* __restrict__ out, int rows, int C)
{
    int t = blockIdx.x * blockDim.x + threadIdx.x;
    if (t >= rows * C) return;
    int r = t / C, c = t % C;
    out[t] = in[(size_t)idx[r] * C + c];
}

__global__ __launch_bounds__(256) void k_maxpool(
    const float* __restrict__ h, const int* __restrict__ nb,
    float* __restrict__ out, int m, int C)
{
    int t = blockIdx.x * blockDim.x + threadIdx.x;
    if (t >= m * C) return;
    int i = t / C, c = t % C;
    float v = -3.0e38f;
#pragma unroll
    for (int k = 0; k < 16; ++k)
        v = fmaxf(v, h[(size_t)nb[i * 16 + k] * C + c]);
    out[t] = v;
}

// rel[i,j,:] = pos[i] - pos[idx(i,j)] with idx(i,16) = i (self loop)
__global__ __launch_bounds__(256) void k_rel(
    const float* __restrict__ pos, const int* __restrict__ nbr,
    float* __restrict__ rel, int n)
{
    int t = blockIdx.x * blockDim.x + threadIdx.x;
    if (t >= n * 17) return;
    int i = t / 17, j = t % 17;
    int nb = (j < 16) ? nbr[i * 16 + j] : i;
    rel[(size_t)t * 3 + 0] = pos[3 * i + 0] - pos[3 * nb + 0];
    rel[(size_t)t * 3 + 1] = pos[3 * i + 1] - pos[3 * nb + 1];
    rel[(size_t)t * 3 + 2] = pos[3 * i + 2] - pos[3 * nb + 2];
}

// t[i,j,c] = a_dst[i,c] - a_src[idx(i,j),c] + delta[i,j,c]
__global__ __launch_bounds__(256) void k_build_t(
    const float* __restrict__ adst, const float* __restrict__ asrc,
    const float* __restrict__ delta, const int* __restrict__ nbr,
    float* __restrict__ tb, int n, int C)
{
    int t = blockIdx.x * blockDim.x + threadIdx.x;
    if (t >= n * 17 * C) return;
    int c = t % C, ij = t / C;
    int i = ij / 17, j = ij % 17;
    int nb = (j < 16) ? nbr[i * 16 + j] : i;
    tb[t] = adst[(size_t)i * C + c] - asrc[(size_t)nb * C + c] + delta[t];
}

// out[i,c] = sum_j softmax_j(alpha[i,j,c]) * (v[idx(i,j),c] + delta[i,j,c])
__global__ __launch_bounds__(256) void k_attn_out(
    const float* __restrict__ alpha, const float* __restrict__ vv,
    const float* __restrict__ delta, const int* __restrict__ nbr,
    float* __restrict__ out, int n, int C)
{
    int t = blockIdx.x * blockDim.x + threadIdx.x;
    if (t >= n * C) return;
    int i = t / C, c = t % C;
    const float* arow = alpha + ((size_t)i * 17) * C + c;
    float mx = -3.0e38f;
#pragma unroll
    for (int j = 0; j < 17; ++j) mx = fmaxf(mx, arow[(size_t)j * C]);
    float den = 0.f, num = 0.f;
#pragma unroll
    for (int j = 0; j < 17; ++j) {
        int nb = (j < 16) ? nbr[i * 16 + j] : i;
        float e = __expf(arow[(size_t)j * C] - mx);
        den += e;
        num += e * (vv[(size_t)nb * C + c] + delta[((size_t)i * 17 + j) * C + c]);
    }
    out[t] = num / den;
}

// knn_interpolate, k=3, inverse-squared-distance weights
__global__ __launch_bounds__(256) void k_interp(
    const float* __restrict__ xsub, const int* __restrict__ nb3,
    const float* __restrict__ poshi, const float* __restrict__ possub,
    float* __restrict__ y, int nhi, int C)
{
    int t = blockIdx.x * blockDim.x + threadIdx.x;
    if (t >= nhi * C) return;
    int i = t / C, c = t % C;
    float px = poshi[3 * i], py = poshi[3 * i + 1], pz = poshi[3 * i + 2];
    float num = 0.f, den = 0.f;
#pragma unroll
    for (int j = 0; j < 3; ++j) {
        int s = nb3[i * 3 + j];
        float dx = px - possub[3 * s], dy = py - possub[3 * s + 1], dz = pz - possub[3 * s + 2];
        float d2 = dx * dx + dy * dy + dz * dz;
        float w = 1.f / fmaxf(d2, 1e-16f);
        num += w * xsub[(size_t)s * C + c];
        den += w;
    }
    y[t] = num / den;
}

// ------------------------------ host side ----------------------------------
static inline int gdiv(int a, int b) { return (a + b - 1) / b; }

static void gemm(hipStream_t s, const float* X, const float* W, const float* b,
                 float* Y, int M, int N, int K, int act)
{
    int tiles = (M >> 4) * (N >> 4);
    k_gemm_wmma<<<gdiv(tiles, 8), 256, 0, s>>>(X, W, b, Y, M, N, K, act);
}
static void lnorm(hipStream_t s, const float* in, float* out, const float* g,
                  const float* b, int n, int C, int relu, const float* add)
{
    k_ln<<<n, 32, 0, s>>>(in, out, g, b, C, relu, add);
}

struct BlockP {
    const float *li_w, *li_b, *lo_w, *lo_b, *g, *b;
    const float *lv_w, *ls_w, *ld_w;
    const float *p0w, *p0b, *p1w, *p1b;
    const float *a0w, *a0b, *a1w, *a1b;
};
struct PW {
    void* const* d; int i;
    const float* next() { return (const float*)d[i++]; }
};
static BlockP popBlock(PW& p)
{
    BlockP B;
    B.li_w = p.next(); B.li_b = p.next();      // lin_in
    B.lo_w = p.next(); B.lo_b = p.next();      // lin_out
    B.g    = p.next(); B.b    = p.next();      // ln_g, ln_b
    B.lv_w = p.next();                         // lin (no bias)
    B.ls_w = p.next();                         // lin_src
    B.ld_w = p.next();                         // lin_dst
    B.p0w = p.next(); B.p0b = p.next();        // pos_nn[0]
    B.p1w = p.next(); B.p1b = p.next();        // pos_nn[1]
    B.a0w = p.next(); B.a0b = p.next();        // attn_nn[0]
    B.a1w = p.next(); B.a1b = p.next();        // attn_nn[1]
    return B;
}

struct Scratch {
    float *h, *adst, *asrc, *vv, *ob;          // [n,C]
    float *rel, *hid, *delta, *tb, *alpha;     // [n*17, ...]
};

static void run_block(hipStream_t s, const BlockP& P, const float* xin, float* xout,
                      const float* pos, const int* knn, int n, int C, const Scratch& sc)
{
    int n17 = n * 17;
    gemm(s, xin, P.li_w, P.li_b, sc.h, n, C, C, 1);                 // relu(lin_in)
    lnorm(s, sc.h, sc.h, P.g, P.b, n, C, 0, nullptr);               // LayerNorm
    gemm(s, sc.h, P.ld_w, nullptr, sc.adst, n, C, C, 0);            // a_dst
    gemm(s, sc.h, P.ls_w, nullptr, sc.asrc, n, C, C, 0);            // a_src
    gemm(s, sc.h, P.lv_w, nullptr, sc.vv,   n, C, C, 0);            // v
    k_rel<<<gdiv(n17, 256), 256, 0, s>>>(pos, knn, sc.rel, n);
    gemm(s, sc.rel, P.p0w, P.p0b, sc.hid,   n17, 64, 3,  1);        // pos_nn[0]
    gemm(s, sc.hid, P.p1w, P.p1b, sc.delta, n17, C,  64, 1);        // pos_nn[1]
    k_build_t<<<gdiv(n17 * C, 256), 256, 0, s>>>(sc.adst, sc.asrc, sc.delta, knn, sc.tb, n, C);
    gemm(s, sc.tb,  P.a0w, P.a0b, sc.hid,   n17, 64, C,  1);        // attn_nn[0]
    gemm(s, sc.hid, P.a1w, P.a1b, sc.alpha, n17, C,  64, 1);        // attn_nn[1]
    k_attn_out<<<gdiv(n * C, 256), 256, 0, s>>>(sc.alpha, sc.vv, sc.delta, knn, sc.ob, n, C);
    gemm(s, sc.ob, P.lo_w, P.lo_b, xout, n, C, C, 1);               // relu(lin_out)
}

extern "C" void kernel_launch(void* const* d_in, const int* in_sizes, int n_in,
                              void* d_out, int out_size, void* d_ws, size_t ws_size,
                              hipStream_t stream)
{
    const int NS[4] = {4096, 1024, 256, 64};
    const int DS[4] = {64, 128, 256, 512};

    const float* x_in   = (const float*)d_in[0];
    const float* pos_in = (const float*)d_in[1];
    // d_in[2] = batch (all zeros, unused)

    PW pw{d_in, 3};
    BlockP inputB = popBlock(pw);
    const float* sm_w = pw.next();
    const float* sm_b = pw.next();
    BlockP summitB = popBlock(pw);
    struct DownP { const float *lw, *lb, *g, *b; BlockP blk; } down[3];
    for (int i = 0; i < 3; ++i) {
        down[i].lw = pw.next(); down[i].lb = pw.next();
        down[i].g  = pw.next(); down[i].b  = pw.next();
        down[i].blk = popBlock(pw);
    }
    struct UpP { const float *subw, *subb, *sg, *sb, *mw, *mb, *g, *b; BlockP blk; } up[3];
    for (int i = 0; i < 3; ++i) {
        up[i].subw = pw.next(); up[i].subb = pw.next();
        up[i].sg   = pw.next(); up[i].sb   = pw.next();
        up[i].mw   = pw.next(); up[i].mb   = pw.next();
        up[i].g    = pw.next(); up[i].b    = pw.next();
        up[i].blk  = popBlock(pw);
    }

    // ---- workspace bump allocator (deterministic layout each call) ----
    char* base = (char*)d_ws;
    size_t off = 0;
    auto alloc_f = [&](size_t n) -> float* {
        float* p = (float*)(base + off);
        off = (off + n * sizeof(float) + 255) & ~(size_t)255;
        return p;
    };
    auto alloc_i = [&](size_t n) -> int* {
        int* p = (int*)(base + off);
        off = (off + n * sizeof(int) + 255) & ~(size_t)255;
        return p;
    };

    const size_t NC_MAX   = 4096 * 64;          // max n*C over all blocks
    const size_t N17_MAX  = (size_t)4096 * 17;
    const size_t N17C_MAX = N17_MAX * 64;       // max n*17*C

    Scratch sc;
    sc.h     = alloc_f(NC_MAX);
    sc.adst  = alloc_f(NC_MAX);
    sc.asrc  = alloc_f(NC_MAX);
    sc.vv    = alloc_f(NC_MAX);
    sc.ob    = alloc_f(NC_MAX);
    sc.rel   = alloc_f(N17_MAX * 3);
    sc.hid   = alloc_f(N17_MAX * 64);
    sc.delta = alloc_f(N17C_MAX);
    sc.tb    = alloc_f(N17C_MAX);
    sc.alpha = alloc_f(N17C_MAX);

    float* xs[4];
    for (int i = 0; i < 4; ++i) xs[i] = alloc_f((size_t)NS[i] * DS[i]);
    const float* pos_l[4];
    pos_l[0] = pos_in;
    float* pos_buf[4] = {nullptr, nullptr, nullptr, nullptr};
    for (int i = 1; i < 4; ++i) { pos_buf[i] = alloc_f((size_t)NS[i] * 3); pos_l[i] = pos_buf[i]; }
    int* knng[4];
    for (int i = 0; i < 4; ++i) knng[i] = alloc_i((size_t)NS[i] * 16);
    int* idc  = alloc_i(1024);
    int* nbuf = alloc_i((size_t)1024 * 16);
    int* nb3  = alloc_i((size_t)4096 * 3);
    float* dn_h  = alloc_f((size_t)4096 * 128);
    float* dn_x  = alloc_f((size_t)1024 * 128);
    float* xbuf0 = alloc_f(NC_MAX);
    float* xbuf1 = alloc_f(NC_MAX);
    float* xsub  = alloc_f((size_t)1024 * 64);
    float* ybuf  = alloc_f(NC_MAX);
    float* xpre  = alloc_f(NC_MAX);
    float* smout = alloc_f((size_t)64 * 512);
    (void)ws_size; (void)in_sizes; (void)n_in;

    // ---- encoder ----
    k_knn<16, true><<<gdiv(NS[0], 256), 256, 0, stream>>>(pos_l[0], pos_l[0], NS[0], NS[0], knng[0]);
    run_block(stream, inputB, x_in, xs[0], pos_l[0], knng[0], NS[0], DS[0], sc);

    for (int i = 0; i < 3; ++i) {
        int n = NS[i], m = NS[i + 1], Ci = DS[i], Co = DS[i + 1];
        gemm(stream, xs[i], down[i].lw, down[i].lb, dn_h, n, Co, Ci, 0);
        lnorm(stream, dn_h, dn_h, down[i].g, down[i].b, n, Co, 1, nullptr);
        k_fps<<<1, 256, 0, stream>>>(pos_l[i], n, m, idc);
        k_gather<<<gdiv(m * 3, 256), 256, 0, stream>>>(pos_l[i], idc, pos_buf[i + 1], m, 3);
        k_knn<16, false><<<gdiv(m, 256), 256, 0, stream>>>(pos_l[i + 1], pos_l[i], m, n, nbuf);
        k_maxpool<<<gdiv(m * Co, 256), 256, 0, stream>>>(dn_h, nbuf, dn_x, m, Co);
        k_knn<16, true><<<gdiv(m, 256), 256, 0, stream>>>(pos_l[i + 1], pos_l[i + 1], m, m, knng[i + 1]);
        run_block(stream, down[i].blk, dn_x, xs[i + 1], pos_l[i + 1], knng[i + 1], m, Co, sc);
    }

    // ---- summit ----
    gemm(stream, xs[3], sm_w, sm_b, smout, NS[3], DS[3], DS[3], 1);
    run_block(stream, summitB, smout, xbuf0, pos_l[3], knng[3], NS[3], DS[3], sc);
    float* xcur = xbuf0;

    // ---- decoder ----
    for (int i = 0; i < 3; ++i) {
        int hi = 2 - i, sub = 3 - i;
        const UpP& U = up[2 - i];
        int nh = NS[hi], ns = NS[sub], Ch = DS[hi], Cs = DS[sub];
        float* xnext = (xcur == xbuf0) ? xbuf1 : xbuf0;

        gemm(stream, xcur, U.subw, U.subb, xsub, ns, Ch, Cs, 0);
        lnorm(stream, xsub, xsub, U.sg, U.sb, ns, Ch, 1, nullptr);
        k_knn<3, false><<<gdiv(nh, 256), 256, 0, stream>>>(pos_l[hi], pos_l[sub], nh, ns, nb3);
        k_interp<<<gdiv(nh * Ch, 256), 256, 0, stream>>>(xsub, nb3, pos_l[hi], pos_l[sub], ybuf, nh, Ch);
        gemm(stream, xs[hi], U.mw, U.mb, xpre, nh, Ch, Ch, 0);
        lnorm(stream, xpre, xpre, U.g, U.b, nh, Ch, 1, ybuf);   // relu(ln(...)) + y
        run_block(stream, U.blk, xpre, xnext, pos_l[hi], knng[hi], nh, Ch, sc);
        xcur = xnext;
    }

    hipMemcpyAsync(d_out, xcur, (size_t)out_size * sizeof(float),
                   hipMemcpyDeviceToDevice, stream);
}
